// MultiHeadAttention_86165633892612
// MI455X (gfx1250) — matrix operand order repaired and checked by reading, not compiled, not run on hardware
//
#include <hip/hip_runtime.h>

// MHA forward for MI455X (gfx1250): bf16 WMMA (16x16x32) with fp32 accum,
// flash-attention style fused softmax, TDM tensor_load_to_lds A-tile staging.
//
// dims
#define SEQ   2048
#define EMB   512
#define HEADS 8
#define HDIM  64
#define NQKV  1536
#define MTOT  4096          // BATCH * SEQ
#define ATT_SCALE 0.125f    // HDIM^-0.5

typedef unsigned short u16;
typedef __attribute__((ext_vector_type(16))) __bf16      v16bf;
typedef __attribute__((ext_vector_type(8)))  float        v8f;
typedef __attribute__((ext_vector_type(4)))  unsigned int u32x4;
typedef __attribute__((ext_vector_type(4)))  int          i32x4;
typedef __attribute__((ext_vector_type(8)))  int          i32x8;

union Frag { v16bf bf; u32x4 q[2]; };

__device__ __forceinline__ u16 f2bf(float f) {
  unsigned u = __float_as_uint(f);
  u += 0x7FFFu + ((u >> 16) & 1u);      // round-to-nearest-even-ish
  return (u16)(u >> 16);
}

__device__ __forceinline__ v8f zero8() {
  v8f z = {0.f, 0.f, 0.f, 0.f, 0.f, 0.f, 0.f, 0.f};
  return z;
}

// --------------------------------------------------------------- TDM stage
// One-shot DMA of a 16x512 bf16 tile (16 KB, rows contiguous, stride 512)
// from global memory into LDS via the Tensor Data Mover. Issue from a single
// wave; TDM ignores EXEC and issues once per wave. Caller barriers afterwards.
__device__ __forceinline__ void tdm_load_tile_16x512(const u16* gsrc, u16* lds_dst) {
#if __has_builtin(__builtin_amdgcn_tensor_load_to_lds)
  unsigned long long ga = (unsigned long long)(const void*)gsrc;
  unsigned lda = (unsigned)(unsigned long long)(void*)lds_dst;  // flat low 32 = LDS offset
  // D# group 0: count=1 | lds_addr | global_addr[56:0] | type=2
  u32x4 g0 = { 1u,
               lda,
               (unsigned)ga,
               (unsigned)((ga >> 32) & 0x01FFFFFFull) | (2u << 30) };
  // D# group 1: data_size=2B; tensor_dim0=512; tensor_dim1=16;
  //             tile_dim0=512; tile_dim1=16; tensor_dim0_stride=512
  i32x8 g1 = { (int)0x00010000u,        // workgroup_mask=0, data_size=1 (2B)
               (int)(512u << 16),       // tensor_dim0[15:0]=512 (in [31:16])
               (int)(16u  << 16),       // tensor_dim0 hi=0 | tensor_dim1[15:0]=16
               (int)(512u << 16),       // tensor_dim1 hi=0 | tile_dim0=512
               16,                      // tile_dim1=16, tile_dim2=0 (unused)
               512,                     // tensor_dim0_stride lo32 = 512
               0, 0 };                  // stride0 hi, tensor_dim1_stride (unused)
  i32x4 gz = {0, 0, 0, 0};              // groups 2/3: tile_dim3/4=0 -> unused
#if __clang_major__ >= 23
  i32x8 gz8 = {0, 0, 0, 0, 0, 0, 0, 0};
  __builtin_amdgcn_tensor_load_to_lds(g0, g1, gz, gz, gz8, 0);
#else
  __builtin_amdgcn_tensor_load_to_lds(g0, g1, gz, gz, 0);
#endif
  __builtin_amdgcn_s_wait_tensorcnt(0);
#else
  (void)gsrc; (void)lds_dst;
#endif
}

// ---------------------------------------------------------------- converts
__global__ void cvt_cast(const float* __restrict__ src, u16* __restrict__ dst, int n) {
  int i = blockIdx.x * blockDim.x + threadIdx.x;
  if (i < n) dst[i] = f2bf(src[i]);
}

// w: [K][N] row-major -> wT: [N][K] row-major (bf16)
__global__ void cvt_transpose(const float* __restrict__ w, u16* __restrict__ wT, int K, int N) {
  int i = blockIdx.x * blockDim.x + threadIdx.x;
  if (i < K * N) {
    int k = i / N, n = i - k * N;
    wT[n * K + k] = f2bf(w[i]);
  }
}

// ---------------------------------------------------------- QKV projection
// C[4096,1536] = Xb[4096,512] @ Wqkv[512,1536]; scatter into per-head
// Q (prescaled), K ([b,h,s,d]) and V transposed ([b,h,d,s]) as bf16.
// grid = (256, 12), block = 256 (8 waves). Each block: 1 m-tile x 8 n-tiles.
__global__ __launch_bounds__(256) void qkv_gemm(
    const u16* __restrict__ xb, const u16* __restrict__ wT,
    u16* __restrict__ qb, u16* __restrict__ kb, u16* __restrict__ vtb) {
  __shared__ u16 As[16 * EMB];                 // 16 KB: one 16 x 512 A tile
  const int mtile = blockIdx.x;
  const int wave  = threadIdx.x >> 5;
  const int lane  = threadIdx.x & 31;
  const int half  = lane >> 4;
  const int lm    = lane & 15;
  const int ntile = blockIdx.y * 8 + wave;     // 0..95

#if __has_builtin(__builtin_amdgcn_tensor_load_to_lds)
  if (threadIdx.x < 32) tdm_load_tile_16x512(xb + mtile * 16 * EMB, As);
#else
  {  // fallback cooperative stage of the A tile (contiguous rows)
    const u32x4* src = (const u32x4*)(xb + mtile * 16 * EMB);
    u32x4* dst = (u32x4*)As;
    for (int t = threadIdx.x; t < 16 * EMB / 8; t += 256) dst[t] = src[t];
  }
#endif
  __syncthreads();

  v8f c = zero8();
  for (int kk = 0; kk < EMB; kk += 32) {
    Frag a, b;
    const u16* ap = As + lm * EMB + kk + 16 * half;
    a.q[0] = *(const u32x4*)(ap);
    a.q[1] = *(const u32x4*)(ap + 8);
    const u16* bp = wT + (ntile * 16 + lm) * EMB + kk + 16 * half;
    b.q[0] = ((const u32x4*)bp)[0];
    b.q[1] = ((const u32x4*)bp)[1];
    c = __builtin_amdgcn_wmma_f32_16x16x32_bf16(false, a.bf, false, b.bf,
                                                (short)0, c, false, false);
  }

  const int gn = ntile * 16 + lm;
#pragma unroll
  for (int r = 0; r < 8; ++r) {
    int gm = mtile * 16 + r + 8 * half;
    int bi = gm >> 11, ss = gm & (SEQ - 1);
    float val = c[r];
    if (gn < EMB) {                                  // Q (fold in scale)
      int hh = gn >> 6, dd = gn & 63;
      qb[((bi * HEADS + hh) * SEQ + ss) * HDIM + dd] = f2bf(val * ATT_SCALE);
    } else if (gn < 2 * EMB) {                       // K
      int g = gn - EMB; int hh = g >> 6, dd = g & 63;
      kb[((bi * HEADS + hh) * SEQ + ss) * HDIM + dd] = f2bf(val);
    } else {                                         // V, stored transposed
      int g = gn - 2 * EMB; int hh = g >> 6, dd = g & 63;
      vtb[((bi * HEADS + hh) * HDIM + dd) * SEQ + ss] = f2bf(val);
    }
  }
}

// -------------------------------------------------------- fused attention
// grid = (16 /*b*h*/, 16), block = 256 (8 waves); each wave owns 16 q rows
// and streams all 2048 keys in blocks of 32 with online softmax.
__global__ __launch_bounds__(256) void attn(
    const u16* __restrict__ qb, const u16* __restrict__ kb,
    const u16* __restrict__ vtb, u16* __restrict__ ctx) {
  __shared__ u16 Plds[8][16][32];              // 8 KB, one 16x32 P tile/wave
  const int bh   = blockIdx.x;
  const int wave = threadIdx.x >> 5;
  const int lane = threadIdx.x & 31;
  const int half = lane >> 4;
  const int lm   = lane & 15;
  const int q0   = (blockIdx.y * 8 + wave) * 16;

  const u16* Q  = qb  + (size_t)bh * SEQ * HDIM;
  const u16* K  = kb  + (size_t)bh * SEQ * HDIM;
  const u16* Vt = vtb + (size_t)bh * HDIM * SEQ;

  Frag aq0, aq1;                               // Q row q0+lm, d 0..31 / 32..63
  {
    const u16* qp = Q + (q0 + lm) * HDIM;
    aq0.q[0] = *(const u32x4*)(qp + 16 * half);
    aq0.q[1] = *(const u32x4*)(qp + 8 + 16 * half);
    aq1.q[0] = *(const u32x4*)(qp + 32 + 16 * half);
    aq1.q[1] = *(const u32x4*)(qp + 40 + 16 * half);
  }

  v8f o0 = zero8(), o1 = zero8(), o2 = zero8(), o3 = zero8();
  float mrun[8], lrun[8];
#pragma unroll
  for (int r = 0; r < 8; ++r) { mrun[r] = -1e30f; lrun[r] = 0.f; }

  for (int j0 = 0; j0 < SEQ; j0 += 32) {
    // S = Q K^T for 32 keys (two 16x16 accum tiles)
    Frag bk;
    v8f s0 = zero8(), s1 = zero8();
    const u16* kpa = K + (j0 + lm) * HDIM + half * 16;
    const u16* kpb = K + (j0 + 16 + lm) * HDIM + half * 16;
    bk.q[0] = ((const u32x4*)kpa)[0]; bk.q[1] = ((const u32x4*)kpa)[1];
    s0 = __builtin_amdgcn_wmma_f32_16x16x32_bf16(false, aq0.bf, false, bk.bf, (short)0, s0, false, false);
    bk.q[0] = ((const u32x4*)(kpa + 32))[0]; bk.q[1] = ((const u32x4*)(kpa + 32))[1];
    s0 = __builtin_amdgcn_wmma_f32_16x16x32_bf16(false, aq1.bf, false, bk.bf, (short)0, s0, false, false);
    bk.q[0] = ((const u32x4*)kpb)[0]; bk.q[1] = ((const u32x4*)kpb)[1];
    s1 = __builtin_amdgcn_wmma_f32_16x16x32_bf16(false, aq0.bf, false, bk.bf, (short)0, s1, false, false);
    bk.q[0] = ((const u32x4*)(kpb + 32))[0]; bk.q[1] = ((const u32x4*)(kpb + 32))[1];
    s1 = __builtin_amdgcn_wmma_f32_16x16x32_bf16(false, aq1.bf, false, bk.bf, (short)0, s1, false, false);

    // online softmax (row = r + 8*half; row spans the 16 lanes of a half)
    float p0[8], p1[8];
#pragma unroll
    for (int r = 0; r < 8; ++r) {
      float v = fmaxf(s0[r], s1[r]);
      v = fmaxf(v, __shfl_xor(v, 8, 32));
      v = fmaxf(v, __shfl_xor(v, 4, 32));
      v = fmaxf(v, __shfl_xor(v, 2, 32));
      v = fmaxf(v, __shfl_xor(v, 1, 32));
      float nmax = fmaxf(mrun[r], v);
      p0[r] = __expf(s0[r] - nmax);
      p1[r] = __expf(s1[r] - nmax);
      float t = p0[r] + p1[r];
      t += __shfl_xor(t, 8, 32);
      t += __shfl_xor(t, 4, 32);
      t += __shfl_xor(t, 2, 32);
      t += __shfl_xor(t, 1, 32);
      float fac = __expf(mrun[r] - nmax);
      lrun[r] = lrun[r] * fac + t;
      mrun[r] = nmax;
      o0[r] *= fac; o1[r] *= fac; o2[r] *= fac; o3[r] *= fac;
    }

    // C-layout -> A-layout for P via per-wave LDS (wave-internal, in order)
#pragma unroll
    for (int r = 0; r < 8; ++r) {
      Plds[wave][r + 8 * half][lm]      = f2bf(p0[r]);
      Plds[wave][r + 8 * half][16 + lm] = f2bf(p1[r]);
    }
    Frag ap;
    ap.q[0] = *(const u32x4*)&Plds[wave][lm][16 * half];
    ap.q[1] = *(const u32x4*)&Plds[wave][lm][8 + 16 * half];

    // O += P @ V  (V^T layout: 16 consecutive keys per lane are contiguous)
    Frag bv;
    const u16* vbase = Vt + j0 + half * 16;
    bv.q[0] = ((const u32x4*)(vbase + (0 * 16 + lm) * SEQ))[0];
    bv.q[1] = ((const u32x4*)(vbase + (0 * 16 + lm) * SEQ))[1];
    o0 = __builtin_amdgcn_wmma_f32_16x16x32_bf16(false, ap.bf, false, bv.bf, (short)0, o0, false, false);
    bv.q[0] = ((const u32x4*)(vbase + (1 * 16 + lm) * SEQ))[0];
    bv.q[1] = ((const u32x4*)(vbase + (1 * 16 + lm) * SEQ))[1];
    o1 = __builtin_amdgcn_wmma_f32_16x16x32_bf16(false, ap.bf, false, bv.bf, (short)0, o1, false, false);
    bv.q[0] = ((const u32x4*)(vbase + (2 * 16 + lm) * SEQ))[0];
    bv.q[1] = ((const u32x4*)(vbase + (2 * 16 + lm) * SEQ))[1];
    o2 = __builtin_amdgcn_wmma_f32_16x16x32_bf16(false, ap.bf, false, bv.bf, (short)0, o2, false, false);
    bv.q[0] = ((const u32x4*)(vbase + (3 * 16 + lm) * SEQ))[0];
    bv.q[1] = ((const u32x4*)(vbase + (3 * 16 + lm) * SEQ))[1];
    o3 = __builtin_amdgcn_wmma_f32_16x16x32_bf16(false, ap.bf, false, bv.bf, (short)0, o3, false, false);
  }

  // epilogue: normalize and pack back to [b, s, e] (bf16 ctx)
  const int b = bh >> 3, h = bh & 7;
#pragma unroll
  for (int r = 0; r < 8; ++r) {
    float inv = 1.0f / lrun[r];
    int row = b * SEQ + q0 + r + 8 * half;
    u16* cp = ctx + (size_t)row * EMB + h * HDIM + lm;
    cp[0]  = f2bf(o0[r] * inv);
    cp[16] = f2bf(o1[r] * inv);
    cp[32] = f2bf(o2[r] * inv);
    cp[48] = f2bf(o3[r] * inv);
  }
}

// --------------------------------------------------------- output GEMM
// out[4096,512] = ctx @ W_o + b_o   (fp32 result). grid = (256, 4).
__global__ __launch_bounds__(256) void out_gemm(
    const u16* __restrict__ ctx, const u16* __restrict__ woT,
    const float* __restrict__ bo, float* __restrict__ out) {
  __shared__ u16 As[16 * EMB];
  const int mtile = blockIdx.x;
  const int wave  = threadIdx.x >> 5;
  const int lane  = threadIdx.x & 31;
  const int half  = lane >> 4;
  const int lm    = lane & 15;
  const int ntile = blockIdx.y * 8 + wave;     // 0..31

#if __has_builtin(__builtin_amdgcn_tensor_load_to_lds)
  if (threadIdx.x < 32) tdm_load_tile_16x512(ctx + mtile * 16 * EMB, As);
#else
  {
    const u32x4* src = (const u32x4*)(ctx + mtile * 16 * EMB);
    u32x4* dst = (u32x4*)As;
    for (int t = threadIdx.x; t < 16 * EMB / 8; t += 256) dst[t] = src[t];
  }
#endif
  __syncthreads();

  v8f c = zero8();
  for (int kk = 0; kk < EMB; kk += 32) {
    Frag a, b;
    const u16* ap = As + lm * EMB + kk + 16 * half;
    a.q[0] = *(const u32x4*)(ap);
    a.q[1] = *(const u32x4*)(ap + 8);
    const u16* bp = woT + (ntile * 16 + lm) * EMB + kk + 16 * half;
    b.q[0] = ((const u32x4*)bp)[0];
    b.q[1] = ((const u32x4*)bp)[1];
    c = __builtin_amdgcn_wmma_f32_16x16x32_bf16(false, a.bf, false, b.bf,
                                                (short)0, c, false, false);
  }

  const int gn = ntile * 16 + lm;
  const float bias = bo[gn];
#pragma unroll
  for (int r = 0; r < 8; ++r) {
    int gm = mtile * 16 + r + 8 * half;
    out[(size_t)gm * EMB + gn] = c[r] + bias;
  }
}

// --------------------------------------------------------------- launcher
extern "C" void kernel_launch(void* const* d_in, const int* in_sizes, int n_in,
                              void* d_out, int out_size, void* d_ws, size_t ws_size,
                              hipStream_t stream) {
  (void)in_sizes; (void)n_in; (void)out_size; (void)ws_size;
  const float* x     = (const float*)d_in[0];
  const float* w_qkv = (const float*)d_in[1];
  const float* w_o   = (const float*)d_in[2];
  const float* b_o   = (const float*)d_in[3];
  float* out = (float*)d_out;
  char*  ws  = (char*)d_ws;

  // workspace layout (bytes), ~23 MB total
  u16* xb    = (u16*)(ws + 0);                 // [4096][512]      4.00 MB
  u16* wqkvT = (u16*)(ws + 4194304);           // [1536][512]      1.50 MB
  u16* woT   = (u16*)(ws + 5767168);           // [512][512]       0.50 MB
  u16* qb    = (u16*)(ws + 6291456);           // [b][h][s][d]     4.00 MB
  u16* kb    = (u16*)(ws + 10485760);          // [b][h][s][d]     4.00 MB
  u16* vtb   = (u16*)(ws + 14680064);          // [b][h][d][s]     4.00 MB
  u16* ctxb  = (u16*)(ws + 18874368);          // [4096][512]      4.00 MB

  cvt_cast<<<(MTOT * EMB + 255) / 256, 256, 0, stream>>>(x, xb, MTOT * EMB);
  cvt_transpose<<<(EMB * NQKV + 255) / 256, 256, 0, stream>>>(w_qkv, wqkvT, EMB, NQKV);
  cvt_transpose<<<(EMB * EMB + 255) / 256, 256, 0, stream>>>(w_o, woT, EMB, EMB);

  qkv_gemm<<<dim3(MTOT / 16, NQKV / 128), 256, 0, stream>>>(xb, wqkvT, qb, kb, vtb);
  attn<<<dim3(2 * HEADS, SEQ / 128), 256, 0, stream>>>(qb, kb, vtb, ctxb);
  out_gemm<<<dim3(MTOT / 16, EMB / 128), 256, 0, stream>>>(ctxb, woT, b_o, out);
}
